// CandidateRefinedMatcher_45715631899334
// MI455X (gfx1250) — matrix operand order repaired and checked by reading, not compiled, not run on hardware
//
#include <hip/hip_runtime.h>
#include <hip/hip_bf16.h>
#include <math.h>

typedef __attribute__((ext_vector_type(16))) _Float16 v16h;
typedef __attribute__((ext_vector_type(8)))  _Float16 v8h;
typedef __attribute__((ext_vector_type(8)))  float    v8f;

#define BB   4
#define NN   3072
#define KC   24
#define DSC  64
#define HID  128
#define WV   4      // waves per block in MLP kernel

__device__ __forceinline__ float nan0f(float x) { return isfinite(x) ? x : 0.0f; }
__device__ __forceinline__ float geluf(float x) {
  return 0.5f * x * (1.0f + erff(x * 0.70710678118654752440f));
}
__device__ __forceinline__ float sigm(float x) { return 1.0f / (1.0f + expf(-x)); }

__device__ __forceinline__ float wredmax(float v) {
  #pragma unroll
  for (int o = 16; o > 0; o >>= 1) v = fmaxf(v, __shfl_xor(v, o, 32));
  return v;
}
__device__ __forceinline__ float wredsum(float v) {
  #pragma unroll
  for (int o = 16; o > 0; o >>= 1) v += __shfl_xor(v, o, 32);
  return v;
}

__device__ __forceinline__ v16h combine16(v8h lo, v8h hi) {
  return __builtin_shufflevector(lo, hi, 0, 1, 2, 3, 4, 5, 6, 7,
                                 8, 9, 10, 11, 12, 13, 14, 15);
}

// Pre-packed B fragment: per (frag, lane) 16 contiguous halves -> 2x b128 loads.
__device__ __forceinline__ v16h loadBpack(const _Float16* __restrict__ P,
                                          int frag, int lane) {
  const v8h* q = (const v8h*)(P + ((size_t)frag * 32 + lane) * 16);
  return combine16(q[0], q[1]);
}

// A-fragment (16-bit 16x32) from a row-major 16x128 f16 LDS tile.
// lane&15 = row; two contiguous 8-half runs: K base..base+7 and base+16..+23.
__device__ __forceinline__ v16h loadA_lds(const _Float16* __restrict__ H,
                                          int kc, int lane) {
  const _Float16* base = H + (lane & 15) * 128 + kc * 32 + ((lane < 16) ? 0 : 8);
  const v8h* p = (const v8h*)base;
  return combine16(p[0], p[2]);          // p[2] == base + 16 halves
}

__device__ __forceinline__ void cvt8(float4 a, float4 b, _Float16* o) {
  o[0] = (_Float16)a.x; o[1] = (_Float16)a.y; o[2] = (_Float16)a.z; o[3] = (_Float16)a.w;
  o[4] = (_Float16)b.x; o[5] = (_Float16)b.y; o[6] = (_Float16)b.z; o[7] = (_Float16)b.w;
}

// 8 consecutive feature columns starting at c0 (multiple of 8), as f16.
__device__ __forceinline__ void feat8(const float* __restrict__ sD,
                                      const float* __restrict__ tailp,
                                      const float* __restrict__ tdrow,
                                      int c0, _Float16* o) {
  if (c0 < 64) {                                    // src descriptor (LDS, f32)
    float4 a = *(const float4*)(sD + c0);
    float4 b = *(const float4*)(sD + c0 + 4);
    cvt8(a, b, o);
  } else if (c0 < 128) {                            // tgt descriptor (global, f32)
    float4 a = *(const float4*)(tdrow + (c0 - 64));
    float4 b = *(const float4*)(tdrow + (c0 - 60));
    cvt8(a, b, o);
  } else if (c0 == 128) {                           // tail cols 128..135
    float4 a = *(const float4*)(tailp);
    float4 b = *(const float4*)(tailp + 4);
    cvt8(a, b, o);
  } else if (c0 == 136) {                           // tail cols 136..139, pad
    float4 a = *(const float4*)(tailp + 8);
    o[0] = (_Float16)a.x; o[1] = (_Float16)a.y;
    o[2] = (_Float16)a.z; o[3] = (_Float16)a.w;
    #pragma unroll
    for (int i = 4; i < 8; ++i) o[i] = (_Float16)0.0f;
  } else {                                          // padded K region
    #pragma unroll
    for (int i = 0; i < 8; ++i) o[i] = (_Float16)0.0f;
  }
}

// ---------------- weight prep: f32 -> f16 packed into WMMA B-fragment order --
__global__ void prep_kernel(const float* __restrict__ W1, const float* __restrict__ W2,
                            const float* __restrict__ Wl, const float* __restrict__ Wo,
                            _Float16* __restrict__ W1p, _Float16* __restrict__ W2p,
                            _Float16* __restrict__ Whp) {
  int stride = gridDim.x * blockDim.x;
  int t = blockIdx.x * blockDim.x + threadIdx.x;
  // layer 1: 5 K-chunks x 8 N-tiles, rows padded 140 -> 160
  for (int i = t; i < 5 * 8 * 32 * 16; i += stride) {
    int e = i & 15, lane = (i >> 4) & 31, nt = (i >> 9) & 7, kc = i >> 12;
    int row = kc * 32 + ((lane < 16) ? 0 : 8) + e + ((e < 8) ? 0 : 8);
    int col = nt * 16 + (lane & 15);
    W1p[i] = (_Float16)((row < 140) ? W1[row * 128 + col] : 0.0f);
  }
  // layer 2: 4 K-chunks x 8 N-tiles
  for (int i = t; i < 4 * 8 * 32 * 16; i += stride) {
    int e = i & 15, lane = (i >> 4) & 31, nt = (i >> 9) & 7, kc = i >> 12;
    int row = kc * 32 + ((lane < 16) ? 0 : 8) + e + ((e < 8) ? 0 : 8);
    int col = nt * 16 + (lane & 15);
    W2p[i] = (_Float16)W2[row * 128 + col];
  }
  // heads: 4 K-chunks x 1 N-tile, cols = [Wl | Wo | 0...]
  for (int i = t; i < 4 * 32 * 16; i += stride) {
    int e = i & 15, lane = (i >> 4) & 31, kc = i >> 9;
    int row = kc * 32 + ((lane < 16) ? 0 : 8) + e + ((e < 8) ? 0 : 8);
    int col = lane & 15;
    float v = 0.0f;
    if (col == 0)     v = Wl[row];
    else if (col < 4) v = Wo[row * 3 + (col - 1)];
    Whp[i] = (_Float16)v;
  }
}

// ---------------- top-K candidate search ----------------
__global__ void __launch_bounds__(128)
topk_kernel(const float* __restrict__ srcc, const float* __restrict__ tgtc,
            int* __restrict__ outIdx, float* __restrict__ outDist) {
  __shared__ float sT[NN * 4];                 // float4-padded tgt cloud (48 KB)
  int b  = blockIdx.x / (NN / 128);
  int n0 = (blockIdx.x % (NN / 128)) * 128;
  const float* tb = tgtc + (size_t)b * NN * 3;
  for (int i = threadIdx.x; i < NN; i += 128) {
    sT[i * 4 + 0] = tb[i * 3 + 0];
    sT[i * 4 + 1] = tb[i * 3 + 1];
    sT[i * 4 + 2] = tb[i * 3 + 2];
    sT[i * 4 + 3] = 0.0f;
  }
  __syncthreads();

  int n = n0 + threadIdx.x;
  const float* sp = srcc + ((size_t)b * NN + n) * 3;
  float sx = sp[0], sy = sp[1], sz = sp[2];

  float bd[KC]; int bi[KC];
  #pragma unroll
  for (int j = 0; j < KC; ++j) { bd[j] = 3.4e38f; bi[j] = 0; }

  for (int m = 0; m < NN; ++m) {
    float4 tp = *(const float4*)(sT + m * 4);   // one ds_load_b128
    float dx = sx - tp.x, dy = sy - tp.y, dz = sz - tp.z;
    float d2 = dx * dx + dy * dy + dz * dz;
    if (d2 < bd[KC - 1]) {
      float cd = d2; int ci = m;
      #pragma unroll
      for (int j = 0; j < KC; ++j) {            // register-resident insert
        if (cd < bd[j]) {
          float td = bd[j]; int ti = bi[j];
          bd[j] = cd; bi[j] = ci; cd = td; ci = ti;
        }
      }
    }
  }
  size_t base = ((size_t)b * NN + n) * KC;
  #pragma unroll
  for (int j = 0; j < KC; ++j) {
    outIdx[base + j]  = bi[j];
    outDist[base + j] = sqrtf(bd[j]);
  }
}

// ---------------- fused MLP + softmax + outputs (one wave per voxel) -------
__global__ void __launch_bounds__(WV * 32)
mlp_kernel(const float* __restrict__ src_can, const float* __restrict__ tgt_can,
           const float* __restrict__ src_desc, const float* __restrict__ tgt_desc,
           const float* __restrict__ src_unc,  const float* __restrict__ tgt_unc,
           const float* __restrict__ sml,      const float* __restrict__ tml,
           const _Float16* __restrict__ W1p, const _Float16* __restrict__ W2p,
           const _Float16* __restrict__ Whp,
           const float* __restrict__ b1, const float* __restrict__ b2,
           const float* __restrict__ blp, const float* __restrict__ bop,
           const int* __restrict__ wIdx, const float* __restrict__ wDist,
           float* __restrict__ out) {
  __shared__ int      sIdx[WV][32];
  __shared__ float    sTail[WV][32][12];
  __shared__ float    sDesc[WV][DSC];
  __shared__ _Float16 sH[WV][16][128];
  __shared__ float    sHead[WV][32][4];

  const int w    = threadIdx.x >> 5;
  const int lane = threadIdx.x & 31;
  const int bn   = blockIdx.x * WV + w;
  const int b    = bn / NN;
  const int n    = bn - b * NN;

  for (int i = lane; i < DSC; i += 32) sDesc[w][i] = src_desc[(size_t)bn * DSC + i];
  int   myIdx  = 0;
  float myDist = 0.0f;
  if (lane < KC) {
    myIdx  = wIdx[(size_t)bn * KC + lane];
    myDist = wDist[(size_t)bn * KC + lane];
  }
  sIdx[w][lane] = myIdx;
  __syncthreads();

  // per-candidate scalar tail (feature cols 128..139)
  {
    float t[12];
    #pragma unroll
    for (int i = 0; i < 12; ++i) t[i] = 0.0f;
    if (lane < KC) {
      int j = myIdx;
      const float* td = tgt_desc + (size_t)(b * NN + j) * DSC;
      float sim = 0.0f;
      for (int d = 0; d < DSC; d += 4) {
        float4 sv = *(const float4*)(&sDesc[w][d]);
        float4 tv = *(const float4*)(td + d);
        sim += sv.x * tv.x + sv.y * tv.y + sv.z * tv.z + sv.w * tv.w;
      }
      float sc0 = src_can[(size_t)bn * 3 + 0], sc1 = src_can[(size_t)bn * 3 + 1],
            sc2 = src_can[(size_t)bn * 3 + 2];
      size_t jb = (size_t)(b * NN + j);
      float tc0 = tgt_can[jb * 3 + 0], tc1 = tgt_can[jb * 3 + 1], tc2 = tgt_can[jb * 3 + 2];
      float srcm = sigm(sml[(size_t)bn * 2 + 0] - sml[(size_t)bn * 2 + 1]);
      float tgtm = sigm(tml[jb * 2 + 0] - tml[jb * 2 + 1]);
      t[0] = sim;
      t[1] = tc0 - sc0; t[2] = tc1 - sc1; t[3] = tc2 - sc2;
      t[4] = (float)(j >> 8) - (float)(n >> 8);
      t[5] = (float)((j >> 4) & 15) - (float)((n >> 4) & 15);
      t[6] = (float)(j & 15) - (float)(n & 15);
      t[7] = myDist;
      t[8] = src_unc[bn];
      t[9] = tgt_unc[jb];
      t[10] = srcm;
      t[11] = tgtm;
    }
    #pragma unroll
    for (int i = 0; i < 12; ++i) sTail[w][lane][i] = t[i];
  }
  __syncthreads();

  for (int mt = 0; mt < 2; ++mt) {
    // ---- build layer-1 A fragments: feats tile 16 x 160 (5 K-chunks) ----
    v16h A1[5];
    {
      int  r   = lane & 15;
      int  cnd = mt * 16 + r;
      bool cv  = cnd < KC;
      int  jc  = sIdx[w][cnd];
      int  kb  = (lane < 16) ? 0 : 8;
      const float* tdrow = tgt_desc + (size_t)(b * NN + jc) * DSC;
      const float* tailp = &sTail[w][cnd][0];
      #pragma unroll
      for (int kc = 0; kc < 5; ++kc) {
        _Float16 tmp[16];
        if (cv) {
          int c0 = kc * 32 + kb;
          feat8(&sDesc[w][0], tailp, tdrow, c0, tmp);
          feat8(&sDesc[w][0], tailp, tdrow, c0 + 16, tmp + 8);
        } else {
          #pragma unroll
          for (int i = 0; i < 16; ++i) tmp[i] = (_Float16)0.0f;
        }
        v16h a;
        #pragma unroll
        for (int i = 0; i < 16; ++i) a[i] = tmp[i];
        A1[kc] = a;
      }
    }
    // ---- layer 1: 5 K-chunks x 8 N-tiles (unroll 2: bound VGPR pressure) ----
    #pragma unroll 2
    for (int nt = 0; nt < 8; ++nt) {
      v8f acc = {0.f, 0.f, 0.f, 0.f, 0.f, 0.f, 0.f, 0.f};
      #pragma unroll
      for (int kc = 0; kc < 5; ++kc) {
        v16h bf = loadBpack(W1p, kc * 8 + nt, lane);
        acc = __builtin_amdgcn_wmma_f32_16x16x32_f16(false, A1[kc], false, bf,
                                                     (short)0, acc, false, false);
      }
      float bias  = b1[nt * 16 + (lane & 15)];
      int   rbase = (lane >> 4) << 3;
      #pragma unroll
      for (int v = 0; v < 8; ++v)
        sH[w][rbase + v][nt * 16 + (lane & 15)] = (_Float16)geluf(acc[v] + bias);
    }
    __syncthreads();

    // ---- layer 2 ----
    v16h A2[4];
    #pragma unroll
    for (int kc = 0; kc < 4; ++kc) A2[kc] = loadA_lds(&sH[w][0][0], kc, lane);
    __syncthreads();
    #pragma unroll 2
    for (int nt = 0; nt < 8; ++nt) {
      v8f acc = {0.f, 0.f, 0.f, 0.f, 0.f, 0.f, 0.f, 0.f};
      #pragma unroll
      for (int kc = 0; kc < 4; ++kc) {
        v16h bf = loadBpack(W2p, kc * 8 + nt, lane);
        acc = __builtin_amdgcn_wmma_f32_16x16x32_f16(false, A2[kc], false, bf,
                                                     (short)0, acc, false, false);
      }
      float bias  = b2[nt * 16 + (lane & 15)];
      int   rbase = (lane >> 4) << 3;
      #pragma unroll
      for (int v = 0; v < 8; ++v)
        sH[w][rbase + v][nt * 16 + (lane & 15)] = (_Float16)geluf(acc[v] + bias);
    }
    __syncthreads();

    // ---- heads: [128 -> 16] (col0 = logit, cols1-3 = offset) ----
    v16h A3[4];
    #pragma unroll
    for (int kc = 0; kc < 4; ++kc) A3[kc] = loadA_lds(&sH[w][0][0], kc, lane);
    v8f acc = {0.f, 0.f, 0.f, 0.f, 0.f, 0.f, 0.f, 0.f};
    #pragma unroll
    for (int kc = 0; kc < 4; ++kc) {
      v16h bf = loadBpack(Whp, kc, lane);
      acc = __builtin_amdgcn_wmma_f32_16x16x32_f16(false, A3[kc], false, bf,
                                                   (short)0, acc, false, false);
    }
    {
      int c     = lane & 15;
      int rbase = (lane >> 4) << 3;
      if (c < 4) {
        #pragma unroll
        for (int v = 0; v < 8; ++v) sHead[w][mt * 16 + rbase + v][c] = acc[v];
      }
    }
    __syncthreads();
  }

  // ---- softmax / stats: lane k owns candidate k (wave32 fits K=24) ----
  int  k   = lane;
  bool act = k < KC;
  float logit = act ? (sHead[w][k][0] + blp[0]) : 0.0f;
  float ox = act ? tanhf(sHead[w][k][1] + bop[0]) : 0.0f;
  float oy = act ? tanhf(sHead[w][k][2] + bop[1]) : 0.0f;
  float oz = act ? tanhf(sHead[w][k][3] + bop[2]) : 0.0f;
  float dist  = act ? sTail[w][k][7] : 1e30f;
  bool  valid = act && ((dist <= 0.45f) || (k == 0));
  float ml = valid ? (logit / 0.07f) : -1e9f;

  float mx   = wredmax(ml);
  float s    = wredsum(expf(ml - mx));
  float logp = ml - mx - logf(s);
  float p    = expf(logp);

  int   j  = sIdx[w][k];
  float cx = act ? ((float)(j >> 8)        + ox) : 0.0f;
  float cy = act ? ((float)((j >> 4) & 15) + oy) : 0.0f;
  float cz = act ? ((float)(j & 15)        + oz) : 0.0f;
  float ex = wredsum(p * cx), ey = wredsum(p * cy), ez = wredsum(p * cz);
  float ent = -wredsum(p * logp);

  float m1 = wredmax(p);
  unsigned long long bal = __ballot(p == m1);
  int   fl = __ffsll((long long)bal) - 1;              // first-occurrence argmax
  float p2 = (lane == fl) ? -1e30f : p;
  float margin  = m1 - wredmax(p2);
  float best_tm = __shfl(sTail[w][k][11], fl, 32);
  float srcm    = sTail[w][0][10];
  float conf    = m1 * sqrtf(fmaxf(srcm * best_tm, 1e-6f));

  float* outE = out;
  float* outR = outE + (size_t)BB * NN * 3;
  float* outP = outR + (size_t)BB * 3 * NN;
  float* outC = outP + (size_t)BB * NN * KC;
  float* outM = outC + (size_t)BB * NN;
  float* outH = outM + (size_t)BB * NN;

  if (act) outP[(size_t)bn * KC + k] = nan0f(p);
  if (lane == 0) {
    outE[(size_t)bn * 3 + 0] = ex;
    outE[(size_t)bn * 3 + 1] = ey;
    outE[(size_t)bn * 3 + 2] = ez;
    float pnx = (float)(n >> 8), pny = (float)((n >> 4) & 15), pnz = (float)(n & 15);
    outR[(size_t)b * 3 * NN + 0 * NN + n] = nan0f(ex - pnx);
    outR[(size_t)b * 3 * NN + 1 * NN + n] = nan0f(ey - pny);
    outR[(size_t)b * 3 * NN + 2 * NN + n] = nan0f(ez - pnz);
    outC[bn] = conf;
    outM[bn] = nan0f(margin);
    outH[bn] = nan0f(ent);
  }
}

// ---------------------------------------------------------------------------
extern "C" void kernel_launch(void* const* d_in, const int* in_sizes, int n_in,
                              void* d_out, int out_size, void* d_ws, size_t ws_size,
                              hipStream_t stream) {
  const float* src_can  = (const float*)d_in[0];
  const float* tgt_can  = (const float*)d_in[1];
  const float* src_desc = (const float*)d_in[2];
  const float* tgt_desc = (const float*)d_in[3];
  const float* src_unc  = (const float*)d_in[4];
  const float* tgt_unc  = (const float*)d_in[5];
  const float* sml      = (const float*)d_in[6];
  const float* tml      = (const float*)d_in[7];
  const float* W1       = (const float*)d_in[8];
  const float* b1       = (const float*)d_in[9];
  const float* W2       = (const float*)d_in[10];
  const float* b2       = (const float*)d_in[11];
  const float* Wl       = (const float*)d_in[12];
  const float* bl       = (const float*)d_in[13];
  const float* Wo       = (const float*)d_in[14];
  const float* bo       = (const float*)d_in[15];

  char* ws = (char*)d_ws;
  size_t off = 0;
  int*      wIdx  = (int*)(ws + off);      off += (size_t)BB * NN * KC * sizeof(int);
  float*    wDist = (float*)(ws + off);    off += (size_t)BB * NN * KC * sizeof(float);
  _Float16* W1p   = (_Float16*)(ws + off); off += (size_t)5 * 8 * 32 * 16 * sizeof(_Float16);
  _Float16* W2p   = (_Float16*)(ws + off); off += (size_t)4 * 8 * 32 * 16 * sizeof(_Float16);
  _Float16* Whp   = (_Float16*)(ws + off); off += (size_t)4 * 32 * 16 * sizeof(_Float16);

  prep_kernel<<<32, 256, 0, stream>>>(W1, W2, Wl, Wo, W1p, W2p, Whp);
  topk_kernel<<<BB * (NN / 128), 128, 0, stream>>>(src_can, tgt_can, wIdx, wDist);
  mlp_kernel<<<(BB * NN) / WV, WV * 32, 0, stream>>>(
      src_can, tgt_can, src_desc, tgt_desc, src_unc, tgt_unc, sml, tml,
      W1p, W2p, Whp, b1, b2, bl, bo, wIdx, wDist, (float*)d_out);
}
